// WrapperYukarinSa_75273596830020
// MI455X (gfx1250) — compile-verified, matrix operand order checked by reading, not run on hardware
//
#include <hip/hip_runtime.h>
#include <hip/hip_bf16.h>

// ---------------------------------------------------------------------------
// YukarinSa forward for MI455X (gfx1250, wave32, WMMA).
//
// Stages:
//   1) prep_kernel    : fp32->bf16 weight conversion + WMMA fragment packing,
//                       state buffers zeroed (deterministic per call).
//   2) encode_gi      : embedding gather -> encoder GEMM (K=288, WMMA bf16)
//                       -> fused input-gate GEMM gi = enc @ W_ih[:, :H]^T
//                       (K=512, N=1536, WMMA bf16), gi stored [t][col][b] bf16
//                       with non-temporal stores (written once, read once).
//   3) gru_kernel     : persistent, 32 WGs x 128 thr. WG s owns hidden
//                       channels [16s,16s+16): W_hh slice staged into LDS via
//                       GLOBAL_LOAD_ASYNC_TO_LDS_B128 (ASYNCcnt), per step one
//                       [64x512]x[512x48] WMMA GEMM with the bf16 hidden state
//                       kept A-fragment-packed in L2 (2x b128 per fragment),
//                       next-step gi prefetched, gate math fp32, per-step
//                       device barrier via counters + s_sleep.
//   4) writeout       : f0[b][t] = post_b + sum of 32 per-WG partials.
// ---------------------------------------------------------------------------

typedef __attribute__((ext_vector_type(16))) __bf16 v16bf;
typedef __attribute__((ext_vector_type(8)))  float  v8f;

#define WMMA_BF16(a, b, c) \
  __builtin_amdgcn_wmma_f32_16x16x32_bf16(false, (a), false, (b), (short)0, (c), false, false)

// problem sizes
constexpr int B_   = 64;
constexpr int L_   = 1024;
constexpr int H_   = 512;
constexpr int CIN  = 276;   // 256 phoneme + 4 accent + 16 speaker
constexpr int G3H  = 1536;  // 3*H
constexpr int NWG  = 32;    // persistent workgroups in gru_kernel

// ---------------- workspace layout (bytes) ----------------
constexpr size_t OFF_ENCWP = 0;                                  // [9kt][32nt][32lane][16e] bf16
constexpr size_t SZ_ENCWP  = (size_t)9 * 32 * 32 * 16 * 2;
constexpr size_t OFF_WIHP  = OFF_ENCWP + SZ_ENCWP;               // [16kt][96nt][32][16] bf16
constexpr size_t SZ_WIHP   = (size_t)16 * 96 * 32 * 16 * 2;
constexpr size_t OFF_WHHP  = OFF_WIHP + SZ_WIHP;                 // [16kt][96nt][32][16] bf16
constexpr size_t SZ_WHHP   = SZ_WIHP;
constexpr size_t OFF_WF0   = OFF_WHHP + SZ_WHHP;                 // [1536] f32 (W_ih[:,512])
constexpr size_t SZ_WF0    = (size_t)G3H * 4;
constexpr size_t OFF_HIDBF = OFF_WF0 + SZ_WF0;                   // [2][4mt][16kt][32][16] bf16 (A-frag packed)
constexpr size_t SZ_HIDBF  = (size_t)2 * 512 * 64 * 2;
constexpr size_t OFF_HID32 = OFF_HIDBF + SZ_HIDBF;               // [2][512][64] f32 (linear)
constexpr size_t SZ_HID32  = (size_t)2 * 512 * 64 * 4;
constexpr size_t OFF_F0P   = OFF_HID32 + SZ_HID32;               // [1025][32][64] f32
constexpr size_t SZ_F0P    = (size_t)1025 * 32 * 64 * 4;
constexpr size_t OFF_CNT   = OFF_F0P + SZ_F0P;                   // [1024] u32 barrier counters
constexpr size_t SZ_CNT    = (size_t)1024 * 4;
constexpr size_t OFF_GI    = (OFF_CNT + SZ_CNT + 255) & ~(size_t)255; // [1024][1536][64] bf16
constexpr size_t SZ_GI     = (size_t)1024 * 1536 * 64 * 2;       // ~201 MB (L2-resident)

// fragment index helper (CDNA5 16-bit WMMA layout, ISA 7.12.2):
//   element e of a lane with hi = lane>>4 maps to k_local = e + (e&8) + hi*8
__device__ __forceinline__ int frag_klocal(int e, int hi) {
  return e + (e & 8) + (hi << 3);
}

// inverse mapping: k_local -> (hi, e, lane-half)
__device__ __forceinline__ void frag_pos(int kl, int& hi, int& e) {
  hi = (kl >> 3) & 1;
  e  = (kl & 7) + (kl >= 16 ? 8 : 0);
}

// ---------------------------------------------------------------------------
// 1) prep: pack weights to bf16 fragments, init state
// ---------------------------------------------------------------------------
__global__ __launch_bounds__(256) void prep_kernel(const float* __restrict__ encW,
                                                   const float* __restrict__ Wih,
                                                   const float* __restrict__ Whh,
                                                   char* __restrict__ ws) {
  __bf16*   encWp = (__bf16*)(ws + OFF_ENCWP);
  __bf16*   Wihp  = (__bf16*)(ws + OFF_WIHP);
  __bf16*   Whhp  = (__bf16*)(ws + OFF_WHHP);
  float*    wf0   = (float*)(ws + OFF_WF0);
  __bf16*   hidbf = (__bf16*)(ws + OFF_HIDBF);
  float*    hid32 = (float*)(ws + OFF_HID32);
  float*    f0p   = (float*)(ws + OFF_F0P);
  unsigned* cnt   = (unsigned*)(ws + OFF_CNT);

  const int T = gridDim.x * blockDim.x;
  const int g = blockIdx.x * blockDim.x + threadIdx.x;

  // encoder B-matrix [K=288 padded][N=512] -> fragments
  for (int i = g; i < 9 * 32 * 512; i += T) {
    int e = i & 15, lane = (i >> 4) & 31, ft = i >> 9;
    int kt = ft >> 5, nt = ft & 31;
    int k = 32 * kt + frag_klocal(e, lane >> 4);
    int n = 16 * nt + (lane & 15);
    encWp[i] = (k < CIN) ? (__bf16)encW[k * H_ + n] : (__bf16)0.0f;
  }
  // gi B-matrix: B[k][n] = W_ih[n][k]  (W_ih is [1536][513])
  for (int i = g; i < 16 * 96 * 512; i += T) {
    int e = i & 15, lane = (i >> 4) & 31, ft = i >> 9;
    int kt = ft / 96, nt = ft % 96;
    int k = 32 * kt + frag_klocal(e, lane >> 4);
    int n = 16 * nt + (lane & 15);
    Wihp[i] = (__bf16)Wih[(size_t)n * (H_ + 1) + k];
  }
  // gh B-matrix: B[k][n] = W_hh[n][k]  (W_hh is [1536][512])
  for (int i = g; i < 16 * 96 * 512; i += T) {
    int e = i & 15, lane = (i >> 4) & 31, ft = i >> 9;
    int kt = ft / 96, nt = ft % 96;
    int k = 32 * kt + frag_klocal(e, lane >> 4);
    int n = 16 * nt + (lane & 15);
    Whhp[i] = (__bf16)Whh[(size_t)n * H_ + k];
  }
  // f0 feedback column of W_ih
  for (int i = g; i < G3H; i += T) wf0[i] = Wih[(size_t)i * (H_ + 1) + H_];
  // zero hidden state (both ping-pong buffers, both precisions)
  for (int i = g; i < 2 * 512 * 64; i += T) {
    hidbf[i] = (__bf16)0.0f;
    hid32[i] = 0.0f;
  }
  // f0 partials slot 0 (f0_prev at t=0 is exactly 0)
  for (int i = g; i < 32 * 64; i += T) f0p[i] = 0.0f;
  // step barrier counters
  for (int i = g; i < 1024; i += T) cnt[i] = 0u;
}

// ---------------------------------------------------------------------------
// 2) fused gather + encoder GEMM + gi GEMM. One block per 16 rows (same batch).
// ---------------------------------------------------------------------------
__global__ __launch_bounds__(256) void encode_gi_kernel(
    const int* __restrict__ vow, const int* __restrict__ con,
    const int* __restrict__ sa, const int* __restrict__ ea,
    const int* __restrict__ sap, const int* __restrict__ eap,
    const int* __restrict__ spk, const float* __restrict__ ph_emb,
    const float* __restrict__ sp_emb, const float* __restrict__ encB,
    const __bf16* __restrict__ encWp, const __bf16* __restrict__ Wihp,
    __bf16* __restrict__ gi) {
  __shared__ __align__(32) __bf16 Ain[9 * 32 * 16];    // gathered input, A-fragment packed (9 KB)
  __shared__ __align__(32) __bf16 Epack[16 * 32 * 16]; // encoder out, A-fragment packed (16 KB)
  __shared__ int   vidx[16], cidx[16];
  __shared__ float accf[16][4];

  const int tid = threadIdx.x;
  const int m0  = blockIdx.x * 16;  // flat row = b*1024 + l
  const int b   = m0 >> 10;
  const int l0  = m0 & 1023;

  if (tid < 16) {
    int ix = b * 1024 + l0 + tid;
    vidx[tid] = vow[ix] + 1;
    cidx[tid] = con[ix] + 1;
    accf[tid][0] = (float)sa[ix];
    accf[tid][1] = (float)ea[ix];
    accf[tid][2] = (float)sap[ix];
    accf[tid][3] = (float)eap[ix];
  }
  __syncthreads();
  const int sid = spk[b];

  // gather 16 rows x 288 channels straight into A-fragment layout
  for (int i = tid; i < 16 * 288; i += 256) {
    int r = i / 288, ch = i % 288;
    float v;
    if (ch < 256) {
      v = ph_emb[vidx[r] * 256 + ch] + ph_emb[cidx[r] * 256 + ch];
    } else if (ch < 260) {
      v = accf[r][ch - 256];
    } else if (ch < CIN) {
      v = sp_emb[sid * 16 + (ch - 260)];
    } else {
      v = 0.0f;  // K padding 276..287
    }
    int kt = ch >> 5, hi, e;
    frag_pos(ch & 31, hi, e);
    Ain[((kt * 32) + hi * 16 + r) * 16 + e] = (__bf16)v;
  }
  __syncthreads();

  const int w = tid >> 5, lane = tid & 31, nl = lane & 15, hi = lane >> 4;

  // phase 1: enc[16][512] = Ain x encW  (each wave: 4 N-tiles, K=288)
  for (int q = 0; q < 4; ++q) {
    int nt = w * 4 + q;
    v8f acc = {0.f, 0.f, 0.f, 0.f, 0.f, 0.f, 0.f, 0.f};
#pragma unroll
    for (int kt = 0; kt < 9; ++kt) {
      v16bf a  = *(const v16bf*)&Ain[(kt * 32 + lane) * 16];
      v16bf bb = *(const v16bf*)&encWp[((size_t)(kt * 32 + nt) * 32 + lane) * 16];
      acc = WMMA_BF16(a, bb, acc);
    }
    int col = 16 * nt + nl;
    float eb = encB[col];
    int kt2 = col >> 5, h2, e2;
    frag_pos(col & 31, h2, e2);
#pragma unroll
    for (int v = 0; v < 8; ++v) {
      int rrow = v + hi * 8;  // row within tile (phase-2 A matrix row)
      Epack[((kt2 * 32) + h2 * 16 + rrow) * 16 + e2] = (__bf16)(acc[v] + eb);
    }
  }
  __syncthreads();

  // phase 2: gi[16][1536] = enc x W_ih_main^T  (each wave: 12 N-tiles, K=512)
  for (int q = 0; q < 12; ++q) {
    int nt = w * 12 + q;
    v8f acc = {0.f, 0.f, 0.f, 0.f, 0.f, 0.f, 0.f, 0.f};
#pragma unroll
    for (int kt = 0; kt < 16; ++kt) {
      v16bf a  = *(const v16bf*)&Epack[(kt * 32 + lane) * 16];
      v16bf bb = *(const v16bf*)&Wihp[((size_t)(kt * 96 + nt) * 32 + lane) * 16];
      acc = WMMA_BF16(a, bb, acc);
    }
    int col = 16 * nt + nl;
#pragma unroll
    for (int v = 0; v < 8; ++v) {
      size_t lt = (size_t)(l0 + v + hi * 8);
      // written once / read once by gru_kernel -> non-temporal store
      __builtin_nontemporal_store((__bf16)acc[v], &gi[(lt * G3H + col) * 64 + b]);
    }
  }
}

// ---------------------------------------------------------------------------
// 3) persistent GRU recurrence
// ---------------------------------------------------------------------------
__device__ __forceinline__ float sigm_f(float x) { return 1.0f / (1.0f + __expf(-x)); }
__device__ __forceinline__ float tanh_f(float x) {
  float t = __expf(-2.0f * fabsf(x));
  float r = (1.0f - t) / (1.0f + t);
  return copysignf(r, x);
}

__global__ __launch_bounds__(128) void gru_kernel(
    const __bf16* __restrict__ gi, const __bf16* __restrict__ Whhp,
    const float* __restrict__ wf0, const float* __restrict__ bih,
    const float* __restrict__ bhh, const float* __restrict__ postW,
    const float* __restrict__ postB, __bf16* __restrict__ hidbf,
    float* __restrict__ hid32, float* __restrict__ f0p, unsigned* __restrict__ cnt) {
  __shared__ __align__(32) __bf16 Bfrag[48 * 512];  // this WG's W_hh slice, fragment packed (48 KB)
  __shared__ float f0_lds[64];

  const int s    = blockIdx.x;         // hidden-channel slice [16s, 16s+16)
  const int tid  = threadIdx.x;
  const int w    = tid >> 5;           // wave -> batch M-tile (rows 16w..16w+15)
  const int lane = tid & 31;
  const int nl   = lane & 15;
  const int hi   = lane >> 4;

  // ---- stage W_hh fragments for gate columns {s, 32+s, 64+s} into LDS via
  //      async Global->LDS DMA (GLOBAL_LOAD_ASYNC_TO_LDS_B128, ASYNCcnt) ----
  for (int i = tid; i < 48 * 64; i += 128) {   // 48 frags x 1KB = 48KB, 16B chunks
    int f  = i >> 6;        // fragment 0..47
    int jj = i & 63;        // 16B chunk within fragment
    int kt = f / 3, g = f % 3;
    unsigned      lds_off = (unsigned)(uintptr_t)&Bfrag[f * 512 + jj * 8];
    const __bf16* gsrc    = &Whhp[((size_t)(kt * 96 + g * 32 + s) * 512) + jj * 8];
    asm volatile("global_load_async_to_lds_b128 %0, %1, off"
                 :: "v"(lds_off), "v"(gsrc) : "memory");
  }
  asm volatile("s_wait_asynccnt 0x0" ::: "memory");

  // loop-invariant per-lane scalars
  const int   c    = 16 * s + nl;       // hidden channel this lane owns
  const float wfr  = wf0[c], wfz = wf0[512 + c], wfn = wf0[1024 + c];
  const float brc  = bih[c] + bhh[c];
  const float bzc  = bih[512 + c] + bhh[512 + c];
  const float binv = bih[1024 + c];
  const float bhnv = bhh[1024 + c];
  const float pw   = postW[c];
  const float pB   = postB[0];
  const int   b0   = 16 * w + hi * 8;   // first of 8 batch rows (D-fragment rows)

  // hidden-state writer mapping: value hid[b][c] lives at fragment position
  //   mt=b>>4, kt=c>>5, lane=hiA*16+(b&15), elem=eA  (A-frag packed layout)
  int hiA, eA;
  frag_pos(c & 31, hiA, eA);
  const int kt_c = c >> 5;
  // element index of (b=b0, c) in the packed buffer; vv advances by 16 elems
  const size_t wbase = (((size_t)w * 16 + kt_c) * 32 + hiA * 16 + hi * 8) * 16 + eA;

  __syncthreads();

  for (int t = 0; t < 1024; ++t) {
    const int cur = t & 1, nxt = cur ^ 1;

    // f0_prev: reduce 32 per-WG partials of the previous step
    if (tid < 64) {
      float acc = (t == 0) ? 0.0f : pB;
      const float* p = f0p + (size_t)t * 2048 + tid;
#pragma unroll 8
      for (int ss = 0; ss < 32; ++ss) acc += p[ss * 64];
      f0_lds[tid] = acc;
    }
    __syncthreads();

    // prefetch next step's gi lines into WGP$ while this step's WMMAs run
    if (t + 1 < 1024) {
      const size_t gn = ((size_t)(t + 1) * G3H) * 64;
      __builtin_prefetch(&gi[gn + (size_t)(c) * 64 + b0], 0, 3);
      __builtin_prefetch(&gi[gn + (size_t)(512 + c) * 64 + b0], 0, 3);
      __builtin_prefetch(&gi[gn + (size_t)(1024 + c) * 64 + b0], 0, 3);
    }

    // gh = hid x W_hh_slice : [16 batch] x [512] x [48 cols]
    // A fragments are pre-packed in global (L2-resident): 2x b128 per fragment
    const __bf16* hc = hidbf + (size_t)cur * 32768 + (size_t)w * 16 * 512;
    v8f ar = {0.f, 0.f, 0.f, 0.f, 0.f, 0.f, 0.f, 0.f};
    v8f az = ar, an = ar;
#pragma unroll
    for (int kt = 0; kt < 16; ++kt) {
      v16bf a   = *(const v16bf*)&hc[((size_t)kt * 32 + lane) * 16];
      v16bf b0f = *(const v16bf*)&Bfrag[((kt * 3 + 0) * 32 + lane) * 16];
      v16bf b1f = *(const v16bf*)&Bfrag[((kt * 3 + 1) * 32 + lane) * 16];
      v16bf b2f = *(const v16bf*)&Bfrag[((kt * 3 + 2) * 32 + lane) * 16];
      ar = WMMA_BF16(a, b0f, ar);
      az = WMMA_BF16(a, b1f, az);
      an = WMMA_BF16(a, b2f, an);
    }

    // gi slices: [t][gate*512 + c][b0..b0+7] -> one b128 per gate
    union { uint4 u; __bf16 e[8]; } Gr, Gz, Gn;
    const size_t gb = ((size_t)t * G3H) * 64;
    Gr.u = *(const uint4*)&gi[gb + (size_t)(c) * 64 + b0];
    Gz.u = *(const uint4*)&gi[gb + (size_t)(512 + c) * 64 + b0];
    Gn.u = *(const uint4*)&gi[gb + (size_t)(1024 + c) * 64 + b0];
    const float* hp = hid32 + (size_t)cur * 32768 + (size_t)c * 64 + b0;

    float hnew[8];
#pragma unroll
    for (int vv = 0; vv < 8; ++vv) {
      float f0pv = f0_lds[b0 + vv];
      float r = sigm_f((float)Gr.e[vv] + wfr * f0pv + brc + ar[vv]);
      float z = sigm_f((float)Gz.e[vv] + wfz * f0pv + bzc + az[vv]);
      float n = tanh_f((float)Gn.e[vv] + wfn * f0pv + binv + r * (an[vv] + bhnv));
      hnew[vv] = (1.0f - z) * n + z * hp[vv];
    }

    // store new hidden state: bf16 copy goes straight into A-fragment layout
    {
      __bf16* ho = hidbf + (size_t)nxt * 32768;
#pragma unroll
      for (int vv = 0; vv < 8; ++vv) ho[wbase + (size_t)vv * 16] = (__bf16)hnew[vv];
      float4 h04 = {hnew[0], hnew[1], hnew[2], hnew[3]};
      float4 h14 = {hnew[4], hnew[5], hnew[6], hnew[7]};
      float* h32o = hid32 + (size_t)nxt * 32768 + (size_t)c * 64 + b0;
      *(float4*)(h32o)     = h04;
      *(float4*)(h32o + 4) = h14;
    }

    // f0 partial: reduce over this WG's 16 channels (lanes within each half)
    float p[8];
#pragma unroll
    for (int vv = 0; vv < 8; ++vv) p[vv] = hnew[vv] * pw;
#pragma unroll
    for (int mask = 1; mask < 16; mask <<= 1) {
#pragma unroll
      for (int vv = 0; vv < 8; ++vv) p[vv] += __shfl_xor(p[vv], mask, 32);
    }
    if (nl == 0) {
      float* outp = f0p + (size_t)(t + 1) * 2048 + s * 64 + b0;
#pragma unroll
      for (int vv = 0; vv < 8; ++vv) outp[vv] = p[vv];
    }

    // device-wide step barrier (counter per step, s_sleep spin)
    __syncthreads();
    if (tid == 0) {
      __threadfence();
      atomicAdd(&cnt[t], 1u);
      while (__hip_atomic_load(&cnt[t], __ATOMIC_RELAXED, __HIP_MEMORY_SCOPE_AGENT) < (unsigned)NWG)
        __builtin_amdgcn_s_sleep(1);
    }
    __syncthreads();
    __threadfence();
  }
}

// ---------------------------------------------------------------------------
// 4) writeout: f0[b][t] = post_b + sum_s partial[t+1][s][b]
// ---------------------------------------------------------------------------
__global__ __launch_bounds__(256) void writeout_kernel(const float* __restrict__ f0p,
                                                       const float* __restrict__ postB,
                                                       float* __restrict__ out) {
  int i = blockIdx.x * blockDim.x + threadIdx.x;
  if (i >= B_ * L_) return;
  int b = i >> 10, t = i & 1023;
  float acc = postB[0];
  const float* p = f0p + (size_t)(t + 1) * 2048 + b;
#pragma unroll 8
  for (int s = 0; s < 32; ++s) acc += p[s * 64];
  out[i] = acc;
}

// ---------------------------------------------------------------------------
extern "C" void kernel_launch(void* const* d_in, const int* in_sizes, int n_in,
                              void* d_out, int out_size, void* d_ws, size_t ws_size,
                              hipStream_t stream) {
  (void)in_sizes; (void)n_in; (void)out_size; (void)ws_size;
  const int*   vow    = (const int*)d_in[1];
  const int*   con    = (const int*)d_in[2];
  const int*   sa     = (const int*)d_in[3];
  const int*   ea     = (const int*)d_in[4];
  const int*   sap    = (const int*)d_in[5];
  const int*   eap    = (const int*)d_in[6];
  const int*   spk    = (const int*)d_in[7];
  const float* ph_emb = (const float*)d_in[8];
  const float* sp_emb = (const float*)d_in[9];
  const float* encW   = (const float*)d_in[10];
  const float* encB   = (const float*)d_in[11];
  const float* Wih    = (const float*)d_in[12];
  const float* bih    = (const float*)d_in[13];
  const float* Whh    = (const float*)d_in[14];
  const float* bhh    = (const float*)d_in[15];
  const float* postW  = (const float*)d_in[16];
  const float* postB  = (const float*)d_in[17];

  char* ws = (char*)d_ws;  // requires ~204 MB scratch (gi tensor is L2-resident)

  prep_kernel<<<1024, 256, 0, stream>>>(encW, Wih, Whh, ws);

  encode_gi_kernel<<<(B_ * L_) / 16, 256, 0, stream>>>(
      vow, con, sa, ea, sap, eap, spk, ph_emb, sp_emb, encB,
      (const __bf16*)(ws + OFF_ENCWP), (const __bf16*)(ws + OFF_WIHP),
      (__bf16*)(ws + OFF_GI));

  gru_kernel<<<NWG, 128, 0, stream>>>(
      (const __bf16*)(ws + OFF_GI), (const __bf16*)(ws + OFF_WHHP),
      (const float*)(ws + OFF_WF0), bih, bhh, postW, postB,
      (__bf16*)(ws + OFF_HIDBF), (float*)(ws + OFF_HID32),
      (float*)(ws + OFF_F0P), (unsigned*)(ws + OFF_CNT));

  writeout_kernel<<<(B_ * L_ + 255) / 256, 256, 0, stream>>>(
      (const float*)(ws + OFF_F0P), postB, (float*)d_out);
}
